// MambaForecaster_54597624267491
// MI455X (gfx1250) — matrix operand order repaired and checked by reading, not compiled, not run on hardware
//
#include <hip/hip_runtime.h>
#include <cstdint>

// ---------------------------------------------------------------------------
// Problem constants (match reference)
// ---------------------------------------------------------------------------
#define BATCH 16
#define SEQL  4096
#define NFEAT 64
#define DMODEL 256
#define DSTATE 16
#define NLAYER 2
#define DCONV 4
#define DINNER 512
#define DTRANK 16
#define NHEADOUT 144          // HOR*NT
#define NTOK (BATCH * SEQL)   // 65536

typedef __attribute__((ext_vector_type(16))) __bf16 v16bf;
typedef __attribute__((ext_vector_type(8)))  __bf16 v8bf;
typedef __attribute__((ext_vector_type(8)))  float  v8f;

union ABFrag { v16bf v; v8bf h[2]; };
union AccU   { v8f v; float f[8]; };

// ---------------------------------------------------------------------------
// fp32 -> bf16 conversion
// ---------------------------------------------------------------------------
__global__ __launch_bounds__(256) void cvt_bf16_kernel(const float* __restrict__ src,
                                                       __bf16* __restrict__ dst,
                                                       size_t n) {
  size_t i = (size_t)blockIdx.x * 256 + threadIdx.x;
  if (i < n) dst[i] = (__bf16)src[i];
}

// ---------------------------------------------------------------------------
// Shared epilogue: D layout -> row = mbase + v, col = bn + 16j + (lane&15)
// ---------------------------------------------------------------------------
template <int EPI>
__device__ __forceinline__ void gemm_epilogue(v8f (&acc)[4], const float* bias,
                                              float* Cf, __bf16* Cb, int M, int N,
                                              int ldc, int bn, int mbase, int ncol) {
#pragma unroll
  for (int j = 0; j < 4; ++j) {
    const int n0 = bn + j * 16;
    if (n0 >= N) continue;                      // uniform guard (N is 16-aligned)
    const int n = n0 + ncol;
    const float bv = (EPI == 0 && bias != nullptr) ? bias[n] : 0.0f;
    AccU uo; uo.v = acc[j];
#pragma unroll
    for (int v = 0; v < 8; ++v) {
      const int row = mbase + v;
      if (row >= M) continue;
      if (EPI == 0)      Cf[(size_t)row * ldc + n] = uo.f[v] + bv;
      else if (EPI == 1) Cb[(size_t)row * ldc + n] = (__bf16)uo.f[v];
      else               Cf[(size_t)row * ldc + n] += uo.f[v];
    }
  }
}

// ---------------------------------------------------------------------------
// Direct-load bf16 WMMA GEMM (branchless inner loop; used for the tiny head
// GEMM). C[M,N] = A[M,K] * W[N,K]^T. K % 32 == 0.
// ---------------------------------------------------------------------------
template <int EPI>
__global__ __launch_bounds__(128)
void wmma_gemm(const __bf16* __restrict__ A, const __bf16* __restrict__ W,
               const float* __restrict__ bias, float* __restrict__ Cf,
               __bf16* __restrict__ Cb, int M, int N, int K, int ldc) {
  const int lane = threadIdx.x & 31;
  const int wave = threadIdx.x >> 5;
  const int bm = blockIdx.x * 64;
  const int bn = blockIdx.y * 64;

  const int mrow = bm + wave * 16 + (lane & 15);
  const int ra   = (mrow < M) ? mrow : (M - 1);      // clamp: loads never OOB
  const int khalfA = (lane >> 4) * 8;
  const int khalfB = (lane >> 4) * 8;
  const int ncol = lane & 15;

  int wrow[4];
#pragma unroll
  for (int j = 0; j < 4; ++j) {
    int n = bn + j * 16 + ncol;
    wrow[j] = (n < N) ? n : (N - 1);
  }

  v8f acc[4] = {v8f{}, v8f{}, v8f{}, v8f{}};

  for (int k0 = 0; k0 < K; k0 += 32) {
    ABFrag au;
    const __bf16* ap = A + (size_t)ra * K + k0 + khalfA;
    au.h[0] = *(const v8bf*)(ap);
    au.h[1] = *(const v8bf*)(ap + 16);
    ABFrag bv[4];
#pragma unroll
    for (int j = 0; j < 4; ++j) {
      const __bf16* wp = W + (size_t)wrow[j] * K + k0 + khalfB;
      bv[j].h[0] = *(const v8bf*)(wp);
      bv[j].h[1] = *(const v8bf*)(wp + 16);
    }
#pragma unroll
    for (int j = 0; j < 4; ++j)
      acc[j] = __builtin_amdgcn_wmma_f32_16x16x32_bf16(
          false, au.v, false, bv[j].v, (short)0, acc[j], false, false);
  }

  const int mbase = bm + wave * 16 + (lane >> 4) * 8;
  gemm_epilogue<EPI>(acc, bias, Cf, Cb, M, N, ldc, bn, mbase, ncol);
}

// ---------------------------------------------------------------------------
// Async-staged bf16 WMMA GEMM (main path). Block tile 128(M) x 64(N), 4 waves;
// each wave owns 32 rows (2 A fragments) so every staged W tile feeds
// 8 WMMAs per k-step. The 64x32 W tile shared by all waves is streamed into
// LDS with global_load_async_to_lds_b128 (ASYNCcnt), double-buffered against
// the WMMAs consuming the previous tile via ds_load_b128. A fragments are
// direct global loads, double-buffered in registers.
// ---------------------------------------------------------------------------
template <int EPI>
__global__ __launch_bounds__(128)
void wmma_gemm_async(const __bf16* __restrict__ A, const __bf16* __restrict__ W,
                     const float* __restrict__ bias, float* __restrict__ Cf,
                     __bf16* __restrict__ Cb, int M, int N, int K, int ldc) {
  __shared__ __align__(32) __bf16 smem[2][64 * 32];   // 2 x 4KB W tiles

  const int tid  = threadIdx.x;
  const int lane = tid & 31;
  const int wave = tid >> 5;
  const int bm = blockIdx.x * 128;
  const int bn = blockIdx.y * 64;

  const int khalfA = (lane >> 4) * 8;
  const int khalfB = (lane >> 4) * 8;
  const int ncol = lane & 15;

  int ra[2];
#pragma unroll
  for (int i = 0; i < 2; ++i) {
    const int mrow = bm + wave * 32 + i * 16 + (lane & 15);
    ra[i] = (mrow < M) ? mrow : (M - 1);
  }

  // Stage piece assignment: 256 x 16B pieces (64 rows x 4 chunks), 2/thread.
  const int p0 = tid * 2;
  const int row0 = p0 >> 2, ch0 = p0 & 3;
  const int row1 = (p0 + 1) >> 2, ch1 = (p0 + 1) & 3;
  const int rn0 = (bn + row0 < N) ? (bn + row0) : (N - 1);
  const int rn1 = (bn + row1 < N) ? (bn + row1) : (N - 1);

  auto stage = [&](int sel, int k0) {
    const uint32_t lbase = (uint32_t)(uintptr_t)(&smem[sel][0]);
    {
      const __bf16* gp = W + (size_t)rn0 * K + k0 + ch0 * 8;
      uint32_t lo = lbase + (uint32_t)((row0 * 32 + ch0 * 8) * 2);
      asm volatile("global_load_async_to_lds_b128 %0, %1, off"
                   :: "v"(lo), "v"((uint64_t)(uintptr_t)gp) : "memory");
    }
    {
      const __bf16* gp = W + (size_t)rn1 * K + k0 + ch1 * 8;
      uint32_t lo = lbase + (uint32_t)((row1 * 32 + ch1 * 8) * 2);
      asm volatile("global_load_async_to_lds_b128 %0, %1, off"
                   :: "v"(lo), "v"((uint64_t)(uintptr_t)gp) : "memory");
    }
  };

  v8f acc[2][4] = {{v8f{}, v8f{}, v8f{}, v8f{}}, {v8f{}, v8f{}, v8f{}, v8f{}}};

  // prologue: stage first W tile, load first A fragments
  stage(0, 0);
  ABFrag a_cur[2];
#pragma unroll
  for (int i = 0; i < 2; ++i) {
    const __bf16* ap = A + (size_t)ra[i] * K + khalfA;
    a_cur[i].h[0] = *(const v8bf*)(ap);
    a_cur[i].h[1] = *(const v8bf*)(ap + 16);
  }
  asm volatile("s_wait_asynccnt 0x0" ::: "memory");
  __syncthreads();

  int sel = 0;
  for (int k0 = 0; k0 < K; k0 += 32) {
    const bool more = (k0 + 32) < K;
    ABFrag a_nxt[2];
    if (more) {
      stage(sel ^ 1, k0 + 32);                       // async: overlaps WMMAs below
#pragma unroll
      for (int i = 0; i < 2; ++i) {
        const __bf16* ap = A + (size_t)ra[i] * K + (k0 + 32) + khalfA;
        a_nxt[i].h[0] = *(const v8bf*)(ap);
        a_nxt[i].h[1] = *(const v8bf*)(ap + 16);
        __builtin_prefetch(ap + 32, 0, 3);           // global_prefetch_b8 (near)
      }
    }
    // gather all four B fragments first, then burst 8 WMMAs
    ABFrag bu[4];
#pragma unroll
    for (int j = 0; j < 4; ++j) {
      const __bf16* lp = &smem[sel][(j * 16 + ncol) * 32 + khalfB];
      bu[j].h[0] = *(const v8bf*)(lp);               // ds_load_b128
      bu[j].h[1] = *(const v8bf*)(lp + 16);
    }
#pragma unroll
    for (int i = 0; i < 2; ++i)
#pragma unroll
      for (int j = 0; j < 4; ++j)
        acc[i][j] = __builtin_amdgcn_wmma_f32_16x16x32_bf16(
            false, a_cur[i].v, false, bu[j].v, (short)0, acc[i][j], false, false);
    if (more) {
      asm volatile("s_wait_asynccnt 0x0" ::: "memory");
      __syncthreads();
      sel ^= 1;
      a_cur[0] = a_nxt[0];
      a_cur[1] = a_nxt[1];
    }
  }

#pragma unroll
  for (int i = 0; i < 2; ++i) {
    const int mbase = bm + wave * 32 + i * 16 + (lane >> 4) * 8;
    gemm_epilogue<EPI>(acc[i], bias, Cf, Cb, M, N, ldc, bn, mbase, ncol);
  }
}

// ---------------------------------------------------------------------------
// LayerNorm over DMODEL per token -> bf16 output
// ---------------------------------------------------------------------------
__global__ __launch_bounds__(256)
void ln_kernel(const float* __restrict__ h, const float* __restrict__ w,
               const float* __restrict__ b, __bf16* __restrict__ u) {
  __shared__ float red[DMODEL];
  const int t = blockIdx.x;
  const int i = threadIdx.x;
  const float x = h[(size_t)t * DMODEL + i];
  red[i] = x;
  __syncthreads();
  for (int s = DMODEL / 2; s > 0; s >>= 1) {
    if (i < s) red[i] += red[i + s];
    __syncthreads();
  }
  const float mu = red[0] * (1.0f / DMODEL);
  __syncthreads();
  const float d = x - mu;
  red[i] = d * d;
  __syncthreads();
  for (int s = DMODEL / 2; s > 0; s >>= 1) {
    if (i < s) red[i] += red[i + s];
    __syncthreads();
  }
  const float var = red[0] * (1.0f / DMODEL);
  const float r = rsqrtf(var + 1e-5f);
  u[(size_t)t * DMODEL + i] = (__bf16)(d * r * w[i] + b[i]);
}

// ---------------------------------------------------------------------------
// Causal depthwise conv1d (kernel DCONV) + bias + SiLU -> xconv bf16
// ---------------------------------------------------------------------------
__global__ __launch_bounds__(256)
void conv_silu_kernel(const __bf16* __restrict__ xz, const float* __restrict__ cw,
                      const float* __restrict__ cb, __bf16* __restrict__ xconv) {
  const size_t idx = (size_t)blockIdx.x * 256 + threadIdx.x;   // over T*DINNER
  if (idx >= (size_t)NTOK * DINNER) return;
  const int d = (int)(idx & (DINNER - 1));
  const size_t t = idx >> 9;            // DINNER == 512
  const int l = (int)(t & (SEQL - 1));
  float acc = cb[d];
#pragma unroll
  for (int j = 0; j < DCONV; ++j) {
    const int lj = l - (DCONV - 1) + j;
    if (lj >= 0) {
      const float xv = (float)xz[(t - (DCONV - 1) + j) * (2 * DINNER) + d];
      acc += cw[d * DCONV + j] * xv;
    }
  }
  const float s = acc / (1.0f + __expf(-acc));
  xconv[t * DINNER + d] = (__bf16)s;
}

// ---------------------------------------------------------------------------
// dt = softplus( xdbl[:, :DTRANK] @ dtw^T + dtb )  -> bf16 [T, DINNER]
// ---------------------------------------------------------------------------
__global__ __launch_bounds__(256)
void dt_proj_kernel(const float* __restrict__ xdbl, const float* __restrict__ dtw,
                    const float* __restrict__ dtb, __bf16* __restrict__ dt) {
  const size_t idx = (size_t)blockIdx.x * 256 + threadIdx.x;
  if (idx >= (size_t)NTOK * DINNER) return;
  const int d = (int)(idx & (DINNER - 1));
  const size_t t = idx >> 9;
  float acc = dtb[d];
  const float* xr = xdbl + t * (DTRANK + 2 * DSTATE);
#pragma unroll
  for (int r = 0; r < DTRANK; ++r) acc += xr[r] * dtw[d * DTRANK + r];
  const float sp = (acc > 20.0f) ? acc : log1pf(__expf(acc));
  dt[t * DINNER + d] = (__bf16)sp;
}

// ---------------------------------------------------------------------------
// Selective scan: state h[DSTATE] in registers, B_t/C_t broadcast via LDS,
// fused D-skip and SiLU(z) gate.
// ---------------------------------------------------------------------------
__global__ __launch_bounds__(128)
void scan_kernel(const __bf16* __restrict__ dt, const __bf16* __restrict__ xconv,
                 const float* __restrict__ xdbl, const __bf16* __restrict__ xz,
                 const float* __restrict__ A_log, const float* __restrict__ Dp,
                 __bf16* __restrict__ y) {
  const int b = blockIdx.x >> 2;
  const int d = ((blockIdx.x & 3) * 128) + threadIdx.x;

  float Arow[DSTATE];
#pragma unroll
  for (int s = 0; s < DSTATE; ++s) Arow[s] = -__expf(A_log[d * DSTATE + s]);
  const float Dd = Dp[d];

  float hs[DSTATE];
#pragma unroll
  for (int s = 0; s < DSTATE; ++s) hs[s] = 0.0f;

  __shared__ float Bs[DSTATE];
  __shared__ float Cs[DSTATE];

  const size_t tbase = (size_t)b * SEQL;
  for (int l = 0; l < SEQL; ++l) {
    const size_t t = tbase + l;
    if (threadIdx.x < 2 * DSTATE) {
      const float* p = xdbl + t * (DTRANK + 2 * DSTATE) + DTRANK;
      if (threadIdx.x < DSTATE) Bs[threadIdx.x] = p[threadIdx.x];
      else                      Cs[threadIdx.x - DSTATE] = p[threadIdx.x];
    }
    __syncthreads();

    const float dtv = (float)dt[t * DINNER + d];
    const float xv  = (float)xconv[t * DINNER + d];
    const float dx  = dtv * xv;
    float yv = 0.0f;
#pragma unroll
    for (int s = 0; s < DSTATE; ++s) {
      hs[s] = hs[s] * __expf(dtv * Arow[s]) + dx * Bs[s];
      yv += hs[s] * Cs[s];
    }
    const float zv = (float)xz[t * (2 * DINNER) + DINNER + d];
    const float g = zv / (1.0f + __expf(-zv));
    y[t * DINNER + d] = (__bf16)((yv + Dd * xv) * g);
    __syncthreads();
  }
}

// ---------------------------------------------------------------------------
// Gather last-token rows of h -> bf16 [BATCH, DMODEL]
// ---------------------------------------------------------------------------
__global__ __launch_bounds__(256)
void gather_last_kernel(const float* __restrict__ h, __bf16* __restrict__ out) {
  const int i = blockIdx.x * 256 + threadIdx.x;   // BATCH*DMODEL = 4096
  const int b = i >> 8;
  const int c = i & 255;
  out[i] = (__bf16)h[((size_t)b * SEQL + (SEQL - 1)) * DMODEL + c];
}

// ---------------------------------------------------------------------------
// Host-side orchestration
// ---------------------------------------------------------------------------
extern "C" void kernel_launch(void* const* d_in, const int* in_sizes, int n_in,
                              void* d_out, int out_size, void* d_ws, size_t ws_size,
                              hipStream_t stream) {
  (void)in_sizes; (void)n_in; (void)out_size; (void)ws_size;

  const float* x      = (const float*)d_in[0];
  const float* ip_w   = (const float*)d_in[1];
  const float* ip_b   = (const float*)d_in[2];
  const float* ln_w   = (const float*)d_in[3];
  const float* ln_b   = (const float*)d_in[4];
  const float* inpw   = (const float*)d_in[5];
  const float* conv_w = (const float*)d_in[6];
  const float* conv_b = (const float*)d_in[7];
  const float* xpw    = (const float*)d_in[8];
  const float* dtw    = (const float*)d_in[9];
  const float* dtb    = (const float*)d_in[10];
  const float* A_log  = (const float*)d_in[11];
  const float* Dp     = (const float*)d_in[12];
  const float* outw   = (const float*)d_in[13];
  const float* op_w   = (const float*)d_in[14];
  const float* op_b   = (const float*)d_in[15];
  float* out = (float*)d_out;

  // workspace carve-out (256B aligned)
  char* base = (char*)d_ws;
  size_t off = 0;
  auto alloc = [&](size_t bytes) -> void* {
    off = (off + 255) & ~(size_t)255;
    void* p = base + off;
    off += bytes;
    return p;
  };
  float*  h_buf   = (float*) alloc((size_t)NTOK * DMODEL * 4);
  __bf16* u_bf    = (__bf16*)alloc((size_t)NTOK * DMODEL * 2);
  __bf16* xz_bf   = (__bf16*)alloc((size_t)NTOK * 2 * DINNER * 2);
  __bf16* xconv   = (__bf16*)alloc((size_t)NTOK * DINNER * 2);
  float*  xdbl    = (float*) alloc((size_t)NTOK * (DTRANK + 2 * DSTATE) * 4);
  __bf16* dt_bf   = (__bf16*)alloc((size_t)NTOK * DINNER * 2);
  __bf16* y_bf    = (__bf16*)alloc((size_t)NTOK * DINNER * 2);
  __bf16* x_bf    = (__bf16*)alloc((size_t)NTOK * NFEAT * 2);
  __bf16* ipw_bf  = (__bf16*)alloc((size_t)DMODEL * NFEAT * 2);
  __bf16* inw_bf  = (__bf16*)alloc((size_t)NLAYER * 2 * DINNER * DMODEL * 2);
  __bf16* xw_bf   = (__bf16*)alloc((size_t)NLAYER * (DTRANK + 2 * DSTATE) * DINNER * 2);
  __bf16* oww_bf  = (__bf16*)alloc((size_t)NLAYER * DMODEL * DINNER * 2);
  __bf16* opw_bf  = (__bf16*)alloc((size_t)NHEADOUT * DMODEL * 2);
  __bf16* hlast   = (__bf16*)alloc((size_t)BATCH * DMODEL * 2);

  auto cvt = [&](const float* src, __bf16* dst, size_t n) {
    cvt_bf16_kernel<<<(unsigned)((n + 255) / 256), 256, 0, stream>>>(src, dst, n);
  };
  // big GEMMs: async-staged W tiles through LDS; block tile 128x64
  auto gemmA = [&](int epi, const __bf16* A, const __bf16* Wb, const float* bias,
                   float* Cf, __bf16* Cb, int M, int N, int K, int ldc) {
    dim3 g((unsigned)((M + 127) / 128), (unsigned)((N + 63) / 64));
    if (epi == 0)      wmma_gemm_async<0><<<g, 128, 0, stream>>>(A, Wb, bias, Cf, Cb, M, N, K, ldc);
    else if (epi == 1) wmma_gemm_async<1><<<g, 128, 0, stream>>>(A, Wb, bias, Cf, Cb, M, N, K, ldc);
    else               wmma_gemm_async<2><<<g, 128, 0, stream>>>(A, Wb, bias, Cf, Cb, M, N, K, ldc);
  };

  // one-time bf16 conversions (weights + input)
  cvt(x,    x_bf,   (size_t)NTOK * NFEAT);
  cvt(ip_w, ipw_bf, (size_t)DMODEL * NFEAT);
  cvt(inpw, inw_bf, (size_t)NLAYER * 2 * DINNER * DMODEL);
  cvt(xpw,  xw_bf,  (size_t)NLAYER * (DTRANK + 2 * DSTATE) * DINNER);
  cvt(outw, oww_bf, (size_t)NLAYER * DMODEL * DINNER);
  cvt(op_w, opw_bf, (size_t)NHEADOUT * DMODEL);

  // h = x @ ip_w^T + ip_b
  gemmA(0, x_bf, ipw_bf, ip_b, h_buf, nullptr, NTOK, DMODEL, NFEAT, DMODEL);

  const size_t elems_di = (size_t)NTOK * DINNER;
  const unsigned grid_di = (unsigned)((elems_di + 255) / 256);

  for (int l = 0; l < NLAYER; ++l) {
    // u = LN(h)
    ln_kernel<<<NTOK, DMODEL, 0, stream>>>(h_buf, ln_w + l * DMODEL, ln_b + l * DMODEL, u_bf);
    // xz = u @ in_proj^T   (x in cols [0,512), z in cols [512,1024))
    gemmA(1, u_bf, inw_bf + (size_t)l * 2 * DINNER * DMODEL, nullptr,
          nullptr, xz_bf, NTOK, 2 * DINNER, DMODEL, 2 * DINNER);
    // causal depthwise conv + SiLU
    conv_silu_kernel<<<grid_di, 256, 0, stream>>>(
        xz_bf, conv_w + (size_t)l * DINNER * DCONV, conv_b + (size_t)l * DINNER, xconv);
    // xdbl = xconv @ x_proj^T   (dt|B|C = 16|16|16 cols)
    gemmA(0, xconv, xw_bf + (size_t)l * (DTRANK + 2 * DSTATE) * DINNER, nullptr,
          xdbl, nullptr, NTOK, DTRANK + 2 * DSTATE, DINNER, DTRANK + 2 * DSTATE);
    // dt = softplus(dt_raw @ dtw^T + dtb)
    dt_proj_kernel<<<grid_di, 256, 0, stream>>>(
        xdbl, dtw + (size_t)l * DINNER * DTRANK, dtb + (size_t)l * DINNER, dt_bf);
    // selective scan (+ D skip, SiLU(z) gate)
    scan_kernel<<<BATCH * (DINNER / 128), 128, 0, stream>>>(
        dt_bf, xconv, xdbl, xz_bf,
        A_log + (size_t)l * DINNER * DSTATE, Dp + (size_t)l * DINNER, y_bf);
    // h += y @ out_proj^T
    gemmA(2, y_bf, oww_bf + (size_t)l * DMODEL * DINNER, nullptr,
          h_buf, nullptr, NTOK, DMODEL, DINNER, DMODEL);
  }

  // head: out = h[:, -1, :] @ op_w^T + op_b  (tiny: direct-load GEMM)
  gather_last_kernel<<<(BATCH * DMODEL) / 256, 256, 0, stream>>>(h_buf, hlast);
  wmma_gemm<0><<<dim3(1, 3), 128, 0, stream>>>(hlast, opw_bf, op_b, out, nullptr,
                                               BATCH, NHEADOUT, DMODEL, NHEADOUT);
}
